// GNNMeshEmbedV2_49744311222714
// MI455X (gfx1250) — matrix operand order repaired and checked by reading, not compiled, not run on hardware
//
#include <hip/hip_runtime.h>

#define NTOT   65536
#define NPER   16384
#define NBATCH 4
#define NEDGE  393216
#define DG     128
#define DOUT   192
#define TOPK   1024

typedef _Float16 v16h __attribute__((ext_vector_type(16)));
typedef _Float16 v8h  __attribute__((ext_vector_type(8)));
typedef float    v8f  __attribute__((ext_vector_type(8)));

// ---------------- workspace layout (bytes) ----------------
#define OFF_H16   (0ull)                        // 65536*128 f16      = 16 MB
#define OFF_ACC   (16777216ull)                 // 65536*128 f32      = 32 MB
#define OFF_CNT   (50331648ull)                 // 65536 f32
#define OFF_H3    (50593792ull)                 // 65536*128 f32      = 32 MB
#define OFF_SREL  (84148224ull)                 // 65536 f32
#define OFF_SROOT (84410368ull)                 // 65536 f32
#define OFF_SACC  (84672512ull)                 // 65536 f32
#define OFF_W1T   (84934656ull)                 // 128*256 f16 (n-major)
#define OFF_W2T   (85000192ull)                 // 128*128 f16
#define OFF_PWT   (85032960ull)                 // 128*128 f16
#define OFF_OWT   (85065728ull)                 // 192*128 f16
#define OFF_PERM  (85114880ull)                 // 4096 i32
#define OFF_TOPV  (85131264ull)                 // 4096 f32

__device__ __forceinline__ v8f wmma16(v16h a, v16h b, v8f c) {
  return __builtin_amdgcn_wmma_f32_16x16x32_f16(false, a, false, b, (short)0, c,
                                                false, false);
}

// A-frag (16x32 f16): lane<16 holds row M=lane, K=k0+0..7 & k0+16..23; lane>=16: +8
__device__ __forceinline__ v16h afrag(const _Float16* rowbase, int k0, int hs) {
  v8h lo = *(const v8h*)(rowbase + k0 + hs * 8);
  v8h hi = *(const v8h*)(rowbase + k0 + 16 + hs * 8);
  return __builtin_shufflevector(lo, hi, 0,1,2,3,4,5,6,7,8,9,10,11,12,13,14,15);
}

// B-frag (32x16 f16): lane<16 holds col N=lane, K=k0+0..7 & k0+16..23; lane>=16: +8
__device__ __forceinline__ v16h bfrag(const _Float16* colbase, int k0, int hs) {
  v8h lo = *(const v8h*)(colbase + k0 + hs * 8);
  v8h hi = *(const v8h*)(colbase + k0 + 16 + hs * 8);
  return __builtin_shufflevector(lo, hi, 0,1,2,3,4,5,6,7,8,9,10,11,12,13,14,15);
}

__device__ __forceinline__ void atomAddF32(float* p, float v) {
  __hip_atomic_fetch_add(p, v, __ATOMIC_RELAXED, __HIP_MEMORY_SCOPE_AGENT);
}

// ---------------- kernel 1: transpose+convert weights to f16 ----------------
__global__ void prep_weights_k(const float* __restrict__ w1, const float* __restrict__ w2,
                               const float* __restrict__ pw, const float* __restrict__ ow,
                               _Float16* W1T, _Float16* W2T, _Float16* PWT, _Float16* OWT) {
  int t = blockIdx.x * 256 + threadIdx.x;
  if (t < 128 * 256) {               // W1T[n*256+k] = w1[k*128+n]
    int n = t >> 8, k = t & 255;
    W1T[t] = (_Float16)w1[k * 128 + n];
  }
  if (t < 128 * 128) {
    int n = t >> 7, k = t & 127;
    W2T[t] = (_Float16)w2[k * 128 + n];
    PWT[t] = (_Float16)pw[k * 128 + n];
  }
  if (t < 192 * 128) {
    int n = t >> 7, k = t & 127;
    OWT[t] = (_Float16)ow[k * 192 + n];
  }
}

// ---------------- kernel 2: node embedding (proj + sincos) -> h f16 ----------------
__global__ void node_embed_k(const float* __restrict__ x, const float* __restrict__ pos,
                             const float* __restrict__ gw, const float* __restrict__ gb,
                             _Float16* __restrict__ h16) {
  int t = blockIdx.x * 256 + threadIdx.x;     // t = node*128 + d
  int node = t >> 7, d = t & 127;
  const float* xr = x + node * 8;
  float acc = gb[d];
#pragma unroll
  for (int j = 0; j < 8; ++j) acc += xr[j] * gw[j * 128 + d];
  float coord = pos[node * 2 + (d >> 6)];
  int kk = d & 63;
  int ki = (kk < 32) ? kk : kk - 32;
  float omega = __powf(10000.0f, -(float)ki * (1.0f / 32.0f));
  float ang = coord * omega;
  acc += (kk < 32) ? __sinf(ang) : __cosf(ang);
  h16[t] = (_Float16)acc;
}

// ---------------- kernel 3: edge MLP + scatter aggregate ----------------
// block = 256 threads = 8 waves; each wave handles 16 edges (full 2-layer MLP via WMMA)
// N dimension is split into passes of 4 WMMA tiles to keep VGPR pressure < 128 (no spills)
__global__ void __launch_bounds__(256)
edge_mlp_k(const _Float16* __restrict__ h16, const int* __restrict__ eidx,
           const _Float16* __restrict__ gW12T,   // W1T then W2T contiguous
           const float* __restrict__ b1, const float* __restrict__ b2,
           float* __restrict__ acc, float* __restrict__ cnt) {
  extern __shared__ char smem[];
  _Float16* sW1T = (_Float16*)smem;             // 32768 halves
  _Float16* sW2T = (_Float16*)(smem + 65536);   // 16384 halves
  _Float16* sM1  = (_Float16*)(smem + 98304);   // 8 waves * 16*128 halves
  int*      sDst = (int*)(smem + 131072);       // 8*16

  { // stage W1T + W2T (98304 B = 6144 uint4)
    const uint4* src = (const uint4*)gW12T;
    uint4* dst = (uint4*)smem;
    for (int i = threadIdx.x; i < 6144; i += 256) dst[i] = src[i];
  }

  const int w = threadIdx.x >> 5, lane = threadIdx.x & 31;
  const int row = lane & 15, hs = lane >> 4;
  const int ebase = (blockIdx.x * 8 + w) * 16;
  const int e = ebase + row;
  const int nsrc = eidx[2 * e], ndst = eidx[2 * e + 1];
  if (hs == 0) sDst[w * 16 + row] = ndst;
  __syncthreads();

  if (hs == 0) atomAddF32(&cnt[ndst], 1.0f);    // one count per edge

  // ---- layer 1: (16x256) @ W1 -> 16x128 ; two passes of 4 N-tiles
#pragma unroll
  for (int half = 0; half < 2; ++half) {
    v8f C[4];
#pragma unroll
    for (int n = 0; n < 4; ++n) { v8f z = {0,0,0,0,0,0,0,0}; C[n] = z; }
#pragma unroll 1
    for (int c = 0; c < 8; ++c) {
      const int node = (c < 4) ? ndst : nsrc;   // concat([h[dst], h[src]])
      v16h a = afrag(h16 + (long long)node * 128, (c & 3) * 32, hs);
#pragma unroll
      for (int n = 0; n < 4; ++n) {
        const int nt = half * 4 + n;
        v16h b = bfrag(sW1T + (nt * 16 + row) * 256, c * 32, hs);
        C[n] = wmma16(a, b, C[n]);
      }
    }
    // bias + exact gelu, store 16x64 f16 sub-tile to LDS for layer 2
#pragma unroll
    for (int n = 0; n < 4; ++n) {
      const int N = (half * 4 + n) * 16 + row;
      const float bv = b1[N];
#pragma unroll
      for (int r = 0; r < 8; ++r) {
        const int M = r + hs * 8;
        float v = C[n][r] + bv;
        v = 0.5f * v * (1.0f + erff(v * 0.70710678118654752f));
        sM1[(w * 16 + M) * 128 + N] = (_Float16)v;
      }
    }
  }
  __syncthreads();

  // ---- layer 2: (16x128) @ W2 -> 16x128 ; two passes of 4 N-tiles
#pragma unroll
  for (int half = 0; half < 2; ++half) {
    v8f D[4];
#pragma unroll
    for (int n = 0; n < 4; ++n) { v8f z = {0,0,0,0,0,0,0,0}; D[n] = z; }
#pragma unroll 1
    for (int c = 0; c < 4; ++c) {
      v16h a = afrag(sM1 + (w * 16 + row) * 128, c * 32, hs);
#pragma unroll
      for (int n = 0; n < 4; ++n) {
        const int nt = half * 4 + n;
        v16h b = bfrag(sW2T + (nt * 16 + row) * 128, c * 32, hs);
        D[n] = wmma16(a, b, D[n]);
      }
    }
    // bias + scatter-add into per-dst accumulator (segment_sum)
#pragma unroll
    for (int n = 0; n < 4; ++n) {
      const int N = (half * 4 + n) * 16 + row;
      const float bv = b2[N];
#pragma unroll
      for (int r = 0; r < 8; ++r) {
        const int M = r + hs * 8;
        const int d2 = sDst[w * 16 + M];
        atomAddF32(&acc[(long long)d2 * 128 + N], D[n][r] + bv);
      }
    }
  }
}

// ---------------- kernel 4: mean + pool_proj (WMMA) ----------------
__global__ void __launch_bounds__(256)
pool_proj_k(const float* __restrict__ accbuf, const float* __restrict__ cnt,
            const _Float16* __restrict__ gPWT, const float* __restrict__ pb,
            float* __restrict__ h3) {
  extern __shared__ char smem[];
  _Float16* sPWT = (_Float16*)smem;             // 16384 halves
  {
    const uint4* src = (const uint4*)gPWT;
    uint4* dst = (uint4*)smem;
    for (int i = threadIdx.x; i < 2048; i += 256) dst[i] = src[i];
  }
  __syncthreads();

  const int w = threadIdx.x >> 5, lane = threadIdx.x & 31;
  const int row = lane & 15, hs = lane >> 4;
  const int nbase = (blockIdx.x * 8 + w) * 16;
  const int node = nbase + row;
  const float inv = 1.0f / fmaxf(cnt[node], 1.0f);

#pragma unroll
  for (int half = 0; half < 2; ++half) {
    v8f C[4];
#pragma unroll
    for (int n = 0; n < 4; ++n) { v8f z = {0,0,0,0,0,0,0,0}; C[n] = z; }
#pragma unroll 1
    for (int c = 0; c < 4; ++c) {
      const int k0 = c * 32;
      const float4 f0 = *(const float4*)(accbuf + (long long)node * 128 + k0 + hs * 8);
      const float4 f1 = *(const float4*)(accbuf + (long long)node * 128 + k0 + hs * 8 + 4);
      const float4 g0 = *(const float4*)(accbuf + (long long)node * 128 + k0 + 16 + hs * 8);
      const float4 g1 = *(const float4*)(accbuf + (long long)node * 128 + k0 + 16 + hs * 8 + 4);
      float t[16] = {f0.x,f0.y,f0.z,f0.w, f1.x,f1.y,f1.z,f1.w,
                     g0.x,g0.y,g0.z,g0.w, g1.x,g1.y,g1.z,g1.w};
      v16h a;
#pragma unroll
      for (int i = 0; i < 16; ++i) a[i] = (_Float16)(t[i] * inv);
#pragma unroll
      for (int n = 0; n < 4; ++n) {
        const int nt = half * 4 + n;
        v16h b = bfrag(sPWT + (nt * 16 + row) * 128, k0, hs);
        C[n] = wmma16(a, b, C[n]);
      }
    }
#pragma unroll
    for (int n = 0; n < 4; ++n) {
      const int N = (half * 4 + n) * 16 + row;
      const float bv = pb[N];
#pragma unroll
      for (int r = 0; r < 8; ++r)
        h3[(long long)(nbase + r + hs * 8) * 128 + N] = C[n][r] + bv;
    }
  }
}

// ---------------- kernel 5: per-node score projections ----------------
__global__ void score_proj_k(const float* __restrict__ h3, const float* __restrict__ wrel,
                             const float* __restrict__ wroot,
                             float* __restrict__ srel, float* __restrict__ sroot) {
  const int node = blockIdx.x * 8 + (threadIdx.x >> 5);
  const int lane = threadIdx.x & 31;
  const float* hr = h3 + (long long)node * 128;
  float s1 = 0.0f, s2 = 0.0f;
#pragma unroll
  for (int d = lane; d < 128; d += 32) {
    const float hv = hr[d];
    s1 += hv * wrel[d];
    s2 += hv * wroot[d];
  }
#pragma unroll
  for (int off = 16; off > 0; off >>= 1) {
    s1 += __shfl_xor(s1, off);
    s2 += __shfl_xor(s2, off);
  }
  if (lane == 0) { srel[node] = s1; sroot[node] = s2; }
}

// ---------------- kernel 6: scalar edge scatter (nbr @ w_rel folded) ----------------
__global__ void edge_score_k(const int* __restrict__ eidx, const float* __restrict__ srel,
                             float* __restrict__ sacc) {
  const int t = blockIdx.x * 256 + threadIdx.x;
  if (t < NEDGE) atomAddF32(&sacc[eidx[2 * t + 1]], srel[eidx[2 * t]]);
}

// ---------------- kernel 7: per-batch full bitonic sort -> top-K ----------------
__global__ void topk_k(const float* __restrict__ sacc, const float* __restrict__ sroot,
                       const float* __restrict__ brel,
                       int* __restrict__ perm, float* __restrict__ topv,
                       float* __restrict__ outb) {
  extern __shared__ char smem[];
  float* skey = (float*)smem;                   // 16384 f32
  int*   sidx = (int*)(smem + 65536);           // 16384 i32
  const int b = blockIdx.x, tid = threadIdx.x;
  const float rb = brel[0];
  for (int j = tid; j < NPER; j += 1024) {
    const int i = b * NPER + j;
    skey[j] = sacc[i] + rb + sroot[i];
    sidx[j] = j;
  }
  __syncthreads();
  for (unsigned k = 2; k <= NPER; k <<= 1) {
    for (unsigned j = k >> 1; j > 0; j >>= 1) {
      for (unsigned i = tid; i < NPER; i += 1024) {
        const unsigned ixj = i ^ j;
        if (ixj > i) {
          const float a = skey[i], c = skey[ixj];
          const bool desc = ((i & k) == 0);
          if (desc ? (a < c) : (a > c)) {
            skey[i] = c; skey[ixj] = a;
            const int t = sidx[i]; sidx[i] = sidx[ixj]; sidx[ixj] = t;
          }
        }
      }
      __syncthreads();
    }
  }
  for (int j = tid; j < TOPK; j += 1024) {
    perm[b * TOPK + j] = b * NPER + sidx[j];
    topv[b * TOPK + j] = skey[j];
    outb[b * TOPK + j] = (float)b;              // batch_pool output
  }
}

// ---------------- kernel 8: gated gather + out_proj (WMMA) ----------------
__global__ void __launch_bounds__(256)
out_proj_k(const float* __restrict__ h3, const int* __restrict__ perm,
           const float* __restrict__ topv, const _Float16* __restrict__ gOWT,
           const float* __restrict__ ob, float* __restrict__ out) {
  extern __shared__ char smem[];
  _Float16* sOWT = (_Float16*)smem;             // 192*128 halves = 49152 B
  {
    const uint4* src = (const uint4*)gOWT;
    uint4* dst = (uint4*)smem;
    for (int i = threadIdx.x; i < 3072; i += 256) dst[i] = src[i];
  }
  __syncthreads();

  const int w = threadIdx.x >> 5, lane = threadIdx.x & 31;
  const int row = lane & 15, hs = lane >> 4;
  const int jbase = (blockIdx.x * 8 + w) * 16;
  const int j = jbase + row;
  const int p = perm[j];
  const float g = tanhf(topv[j]);

#pragma unroll 1
  for (int third = 0; third < 3; ++third) {     // 12 N-tiles in 3 passes of 4
    v8f C[4];
#pragma unroll
    for (int n = 0; n < 4; ++n) { v8f z = {0,0,0,0,0,0,0,0}; C[n] = z; }
#pragma unroll 1
    for (int c = 0; c < 4; ++c) {
      const int k0 = c * 32;
      const float4 f0 = *(const float4*)(h3 + (long long)p * 128 + k0 + hs * 8);
      const float4 f1 = *(const float4*)(h3 + (long long)p * 128 + k0 + hs * 8 + 4);
      const float4 g0 = *(const float4*)(h3 + (long long)p * 128 + k0 + 16 + hs * 8);
      const float4 g1 = *(const float4*)(h3 + (long long)p * 128 + k0 + 16 + hs * 8 + 4);
      float t[16] = {f0.x,f0.y,f0.z,f0.w, f1.x,f1.y,f1.z,f1.w,
                     g0.x,g0.y,g0.z,g0.w, g1.x,g1.y,g1.z,g1.w};
      v16h a;
#pragma unroll
      for (int i = 0; i < 16; ++i) a[i] = (_Float16)(t[i] * g);
#pragma unroll
      for (int n = 0; n < 4; ++n) {
        const int nt = third * 4 + n;
        v16h b = bfrag(sOWT + (nt * 16 + row) * 128, k0, hs);
        C[n] = wmma16(a, b, C[n]);
      }
    }
#pragma unroll
    for (int n = 0; n < 4; ++n) {
      const int N = (third * 4 + n) * 16 + row;
      const float bv = ob[N];
#pragma unroll
      for (int r = 0; r < 8; ++r)
        out[(long long)(jbase + r + hs * 8) * DOUT + N] = C[n][r] + bv;
    }
  }
}

// ---------------- host launcher ----------------
extern "C" void kernel_launch(void* const* d_in, const int* in_sizes, int n_in,
                              void* d_out, int out_size, void* d_ws, size_t ws_size,
                              hipStream_t stream) {
  const float* x     = (const float*)d_in[0];
  const float* pos   = (const float*)d_in[1];
  const int*   eidx  = (const int*)d_in[2];
  const float* gw    = (const float*)d_in[4];
  const float* gb    = (const float*)d_in[5];
  const float* w1    = (const float*)d_in[6];
  const float* b1    = (const float*)d_in[7];
  const float* w2    = (const float*)d_in[8];
  const float* b2    = (const float*)d_in[9];
  const float* pw    = (const float*)d_in[10];
  const float* pb    = (const float*)d_in[11];
  const float* wrel  = (const float*)d_in[12];
  const float* brel  = (const float*)d_in[13];
  const float* wroot = (const float*)d_in[14];
  const float* ow    = (const float*)d_in[15];
  const float* ob    = (const float*)d_in[16];

  char* ws = (char*)d_ws;
  _Float16* h16  = (_Float16*)(ws + OFF_H16);
  float*    acc  = (float*)(ws + OFF_ACC);
  float*    cnt  = (float*)(ws + OFF_CNT);
  float*    h3   = (float*)(ws + OFF_H3);
  float*    srel = (float*)(ws + OFF_SREL);
  float*    srt  = (float*)(ws + OFF_SROOT);
  float*    sacc = (float*)(ws + OFF_SACC);
  _Float16* W1T  = (_Float16*)(ws + OFF_W1T);
  _Float16* W2T  = (_Float16*)(ws + OFF_W2T);
  _Float16* PWT  = (_Float16*)(ws + OFF_PWT);
  _Float16* OWT  = (_Float16*)(ws + OFF_OWT);
  int*      perm = (int*)(ws + OFF_PERM);
  float*    topv = (float*)(ws + OFF_TOPV);

  float* out_pool  = (float*)d_out;
  float* out_batch = out_pool + (size_t)NBATCH * TOPK * DOUT;

  // zero: acc..cnt is contiguous; sacc separate
  hipMemsetAsync(ws + OFF_ACC, 0, (size_t)NTOT * DG * 4 + (size_t)NTOT * 4, stream);
  hipMemsetAsync(ws + OFF_SACC, 0, (size_t)NTOT * 4, stream);

  prep_weights_k<<<128, 256, 0, stream>>>(w1, w2, pw, ow, W1T, W2T, PWT, OWT);
  node_embed_k<<<(NTOT * DG) / 256, 256, 0, stream>>>(x, pos, gw, gb, h16);
  edge_mlp_k<<<NEDGE / 128, 256, 131584, stream>>>(h16, eidx, W1T, b1, b2, acc, cnt);
  pool_proj_k<<<NTOT / 128, 256, 32768, stream>>>(acc, cnt, PWT, pb, h3);
  score_proj_k<<<NTOT / 8, 256, 0, stream>>>(h3, wrel, wroot, srel, srt);
  edge_score_k<<<(NEDGE + 255) / 256, 256, 0, stream>>>(eidx, srel, sacc);
  topk_k<<<NBATCH, 1024, 131072, stream>>>(sacc, srt, brel, perm, topv, out_batch);
  out_proj_k<<<(NBATCH * TOPK) / 128, 256, 49152, stream>>>(h3, perm, topv, OWT, ob, out_pool);
}
